// MinGRU_7773890806204
// MI455X (gfx1250) — matrix operand order repaired and checked by reading, not compiled, not run on hardware
//
#include <hip/hip_runtime.h>

#define T_SEQ 512
#define B_SZ  64
#define DH    512
#define BH    (B_SZ * DH)   // 32768 elements of one hidden state

typedef unsigned short u16;
typedef __attribute__((ext_vector_type(16))) __bf16 v16bf;
typedef __attribute__((ext_vector_type(8)))  float  v8f;

union FragBF { uint4 u[2]; v16bf v; };

__device__ __forceinline__ u16 f2bf(float f) {
    unsigned u = __float_as_uint(f);
    unsigned r = u + 0x7fffu + ((u >> 16) & 1u);   // round-to-nearest-even
    return (u16)(r >> 16);
}

__device__ __forceinline__ float fast_tanh(float x) {
#if __has_builtin(__builtin_amdgcn_tanhf)
    return __builtin_amdgcn_tanhf(x);
#else
    return tanhf(x);
#endif
}
__device__ __forceinline__ float fast_sigmoid(float x) {
    return 0.5f * fast_tanh(0.5f * x) + 0.5f;      // exact identity
}

// ---------------- grid-wide barrier (persistent cooperative kernel) --------
struct GridBar { unsigned arrive; unsigned phase; unsigned pad[14]; };

__device__ __forceinline__ void grid_sync(GridBar* b, unsigned nblk) {
    __syncthreads();
    if (threadIdx.x == 0) {
        __threadfence();                                   // release my stores
        unsigned ph = __atomic_load_n(&b->phase, __ATOMIC_RELAXED);
        if (atomicAdd(&b->arrive, 1u) == nblk - 1u) {
            b->arrive = 0u;
            __threadfence();
            atomicAdd(&b->phase, 1u);
        } else {
            while (__atomic_load_n(&b->phase, __ATOMIC_RELAXED) == ph)
                __builtin_amdgcn_s_sleep(2);
        }
        __threadfence();                                   // acquire
    }
    __syncthreads();
}

// ---------------- dual-output K=1024 WMMA tile GEMM ------------------------
// A rows (16x32 bf16 frag): lane holds row M=lane%16, K chunks at (lane/16)*8 and +16.
// B rows from W^T [N][K] bf16: lane holds col N=lane%16, K chunks at (lane/16)*8 and +16.
__device__ __forceinline__ void gemm_zh(
    const u16* __restrict__ p0, const u16* __restrict__ p1,   // per-lane row ptrs, 512 K each
    const u16* __restrict__ wzrow, const u16* __restrict__ whrow, // per-lane B row ptrs, K=1024
    int lane, v8f& accz, v8f& acch)
{
    const int ka = (lane >> 4) << 3;   // element offset 0 or 8
    const int kb = (lane >> 4) << 3;   // element offset 0 or 8
    accz = v8f{0.f,0.f,0.f,0.f,0.f,0.f,0.f,0.f};
    acch = v8f{0.f,0.f,0.f,0.f,0.f,0.f,0.f,0.f};
#pragma unroll
    for (int kt = 0; kt < 32; ++kt) {
        const u16* ap = (kt < 16) ? (p0 + kt * 32 + ka) : (p1 + (kt - 16) * 32 + ka);
        FragBF A, Bz, Bh;
        A.u[0]  = *(const uint4*)(ap);
        A.u[1]  = *(const uint4*)(ap + 16);
        const u16* zp = wzrow + kt * 32 + kb;
        Bz.u[0] = *(const uint4*)(zp);
        Bz.u[1] = *(const uint4*)(zp + 16);
        const u16* hp = whrow + kt * 32 + kb;
        Bh.u[0] = *(const uint4*)(hp);
        Bh.u[1] = *(const uint4*)(hp + 16);
        accz = __builtin_amdgcn_wmma_f32_16x16x32_bf16(false, A.v, false, Bz.v,
                                                       (short)0, accz, false, false);
        acch = __builtin_amdgcn_wmma_f32_16x16x32_bf16(false, A.v, false, Bh.v,
                                                       (short)0, acch, false, false);
    }
}

// ---------------- persistent pipelined recurrence --------------------------
// 32 blocks x 256 thr = 256 waves. Waves [0,128): layer0 @ t=s ; [128,256): layer1 @ t=s-1.
// One grid barrier per stage, 513 stages total.
__global__ void __launch_bounds__(256, 1) k_recurrence(
    const u16* __restrict__ xb,                                 // [B,T,512] bf16
    const u16* __restrict__ wz0t, const u16* __restrict__ wh0t, // [512][1024] bf16 (N-major)
    const u16* __restrict__ wz1t, const u16* __restrict__ wh1t,
    const float* __restrict__ bz0, const float* __restrict__ bh0,
    const float* __restrict__ bz1, const float* __restrict__ bh1,
    float* __restrict__ h0f, u16* __restrict__ h0b,             // h0b: [2][BH] (t-parity)
    float* __restrict__ h1f, u16* __restrict__ h1b,             // h1b: [2][BH]
    u16* __restrict__ ysb,                                      // [B,T,512] bf16
    GridBar* bar, int nblk)
{
    const int lane = threadIdx.x & 31;
    const int wave = (blockIdx.x << 3) + (threadIdx.x >> 5);
    const int grp  = wave >> 7;          // 0 = layer0, 1 = layer1
    const int job  = wave & 127;         // 4 M-tiles x 32 N-tiles
    const int tm   = job >> 5;
    const int tn   = job & 31;
    const int col  = (tn << 4) + (lane & 15);            // output column (N)
    const int rowm = (tm << 4) + (lane & 15);            // A-fragment row (batch b)
    const int mb   = (tm << 4) + ((lane >> 4) << 3);     // C-fragment row base
    const int idx0 = mb * DH + col;

    const u16* wzrow; const u16* whrow; float bzv, bhv;
    if (grp == 0) { wzrow = wz0t + col * 1024; whrow = wh0t + col * 1024;
                    bzv = bz0[col]; bhv = bh0[col]; }
    else          { wzrow = wz1t + col * 1024; whrow = wh1t + col * 1024;
                    bzv = bz1[col]; bhv = bh1[col]; }

    for (int s = 0; s <= T_SEQ; ++s) {
        if (grp == 0) {
            if (s < T_SEQ) {
                const int t = s;
                const u16* p0 = xb  + (rowm * T_SEQ + t) * DH;           // x_t
                const u16* p1 = h0b + ((t + 1) & 1) * BH + rowm * DH;    // h0[t-1]
                v8f az, ah;
                gemm_zh(p0, p1, wzrow, whrow, lane, az, ah);
                u16* hbo = h0b + (t & 1) * BH;
#pragma unroll
                for (int r = 0; r < 8; ++r) {
                    float z  = fast_sigmoid(az[r] + bzv);
                    float hc = fast_tanh(ah[r] + bhv);
                    float ho = h0f[idx0 + r * DH];
                    float hn = __builtin_fmaf(z, hc - ho, ho);
                    h0f[idx0 + r * DH] = hn;
                    hbo[idx0 + r * DH] = f2bf(hn);
                }
            }
        } else {
            if (s >= 1) {
                const int t = s - 1;
                const u16* p0 = h0b + (t & 1) * BH + rowm * DH;          // h0[t]
                const u16* p1 = h1b + ((t + 1) & 1) * BH + rowm * DH;    // h1[t-1]
                v8f az, ah;
                gemm_zh(p0, p1, wzrow, whrow, lane, az, ah);
                u16* hbo = h1b + (t & 1) * BH;
#pragma unroll
                for (int r = 0; r < 8; ++r) {
                    float z  = fast_sigmoid(az[r] + bzv);
                    float hc = fast_tanh(ah[r] + bhv);
                    float ho = h1f[idx0 + r * DH];
                    float hn = __builtin_fmaf(z, hc - ho, ho);
                    h1f[idx0 + r * DH] = hn;
                    hbo[idx0 + r * DH] = f2bf(hn);
                    ysb[((mb + r) * T_SEQ + t) * DH + col] = f2bf(hn);
                }
            }
        }
        grid_sync(bar, (unsigned)nblk);
    }
}

// ---------------- output projection: [32768,512] x [512,512] ---------------
__global__ void __launch_bounds__(256) k_proj(
    const u16* __restrict__ ysb, const u16* __restrict__ wot,   // wot: [512][512] N-major
    const float* __restrict__ bo, float* __restrict__ out)
{
    const int lane = threadIdx.x & 31;
    const int wave = (blockIdx.x << 3) + (threadIdx.x >> 5);    // 65536 waves
    const int tm = wave >> 5;                                   // 2048 M-tiles
    const int tn = wave & 31;
    const int col  = (tn << 4) + (lane & 15);
    const int rowm = (tm << 4) + (lane & 15);
    const u16* arow = ysb + rowm * DH;
    const u16* brow = wot + col * DH;
    const int ka = (lane >> 4) << 3;
    const int kb = (lane >> 4) << 3;
    v8f acc = v8f{0.f,0.f,0.f,0.f,0.f,0.f,0.f,0.f};
#pragma unroll
    for (int kt = 0; kt < 16; ++kt) {
        FragBF A, Bw;
        const u16* ap = arow + kt * 32 + ka;
        A.u[0]  = *(const uint4*)(ap);
        A.u[1]  = *(const uint4*)(ap + 16);
        const u16* bp = brow + kt * 32 + kb;
        Bw.u[0] = *(const uint4*)(bp);
        Bw.u[1] = *(const uint4*)(bp + 16);
        acc = __builtin_amdgcn_wmma_f32_16x16x32_bf16(false, A.v, false, Bw.v,
                                                      (short)0, acc, false, false);
    }
    const float bv = bo[col];
    const int mb = (tm << 4) + ((lane >> 4) << 3);
#pragma unroll
    for (int r = 0; r < 8; ++r)
        out[(mb + r) * DH + col] = acc[r] + bv;
}

// ---------------- format conversion / init helpers -------------------------
__global__ void k_cvt_w(const float* __restrict__ src, u16* __restrict__ dst,
                        int K, int N) {                  // src [K][N] -> dst [N][K] bf16
    int total = K * N;
    for (int i = blockIdx.x * blockDim.x + threadIdx.x; i < total;
         i += blockDim.x * gridDim.x) {
        int n = i / K, k = i - n * K;
        dst[i] = f2bf(src[k * N + n]);
    }
}

__global__ void k_cvt_x(const float* __restrict__ src, u16* __restrict__ dst, int total) {
    for (int i = blockIdx.x * blockDim.x + threadIdx.x; i < total;
         i += blockDim.x * gridDim.x)
        dst[i] = f2bf(src[i]);
}

__global__ void k_zero(unsigned* __restrict__ p, int nwords) {
    for (int i = blockIdx.x * blockDim.x + threadIdx.x; i < nwords;
         i += blockDim.x * gridDim.x)
        p[i] = 0u;
}

__global__ void k_copy_h(const float* __restrict__ h0f, const float* __restrict__ h1f,
                         float* __restrict__ dst) {
    int i = blockIdx.x * blockDim.x + threadIdx.x;
    if (i < BH) { dst[i] = h0f[i]; dst[BH + i] = h1f[i]; }
}

// ---------------- launch ---------------------------------------------------
extern "C" void kernel_launch(void* const* d_in, const int* in_sizes, int n_in,
                              void* d_out, int out_size, void* d_ws, size_t ws_size,
                              hipStream_t stream)
{
    const float* x   = (const float*)d_in[0];
    const float* Wz0 = (const float*)d_in[1];
    const float* bz0 = (const float*)d_in[2];
    const float* Wh0 = (const float*)d_in[3];
    const float* bh0 = (const float*)d_in[4];
    const float* Wz1 = (const float*)d_in[5];
    const float* bz1 = (const float*)d_in[6];
    const float* Wh1 = (const float*)d_in[7];
    const float* bh1 = (const float*)d_in[8];
    const float* Wo  = (const float*)d_in[9];
    const float* bo  = (const float*)d_in[10];
    float* out = (float*)d_out;

    char* w = (char*)d_ws;
    u16* wz0t = (u16*)(w + 0x000000);      // 1 MiB each (bf16 [512][1024])
    u16* wh0t = (u16*)(w + 0x100000);
    u16* wz1t = (u16*)(w + 0x200000);
    u16* wh1t = (u16*)(w + 0x300000);
    u16* wot  = (u16*)(w + 0x400000);      // 512 KiB (bf16 [512][512])
    u16* xb   = (u16*)(w + 0x480000);      // 32 MiB bf16 x
    u16* ysb  = (u16*)(w + 0x2480000);     // 32 MiB bf16 ys
    float* h0f = (float*)(w + 0x4480000);  // carries + barrier (zeroed each launch)
    float* h1f = h0f + BH;
    u16*   h0b = (u16*)(h1f + BH);
    u16*   h1b = h0b + 2 * BH;
    GridBar* bar = (GridBar*)(h1b + 2 * BH);

    k_cvt_w<<<512, 256, 0, stream>>>(Wz0, wz0t, 1024, 512);
    k_cvt_w<<<512, 256, 0, stream>>>(Wh0, wh0t, 1024, 512);
    k_cvt_w<<<512, 256, 0, stream>>>(Wz1, wz1t, 1024, 512);
    k_cvt_w<<<512, 256, 0, stream>>>(Wh1, wh1t, 1024, 512);
    k_cvt_w<<<512, 256, 0, stream>>>(Wo,  wot,  512,  512);
    k_cvt_x<<<4096, 256, 0, stream>>>(x, xb, B_SZ * T_SEQ * DH);

    const int zero_words = (2 * BH * 4 + 2 * (2 * BH) * 2 + (int)sizeof(GridBar)) / 4;
    k_zero<<<128, 256, 0, stream>>>((unsigned*)h0f, zero_words);

    const int NBLK = 32;  // 256 waves, all co-resident; one tile-job per wave per layer
    k_recurrence<<<NBLK, 256, 0, stream>>>(xb, wz0t, wh0t, wz1t, wh1t,
                                           bz0, bh0, bz1, bh1,
                                           h0f, h0b, h1f, h1b, ysb, bar, NBLK);

    k_proj<<<8192, 256, 0, stream>>>(ysb, wot, bo, out);
    k_copy_h<<<128, 256, 0, stream>>>(h0f, h1f, out + (size_t)B_SZ * T_SEQ * DH);
}
